// Trittention_8899172237948
// MI455X (gfx1250) — matrix operand order repaired and checked by reading, not compiled, not run on hardware
//
#include <hip/hip_runtime.h>
#include <hip/hip_bf16.h>
#include <math.h>

typedef __attribute__((ext_vector_type(16))) __bf16 v16bf;
typedef __attribute__((ext_vector_type(8)))  float  v8f;

#define SDIM 192
#define DDIM 64
#define TILES 12              // SDIM/16
#define NTILES (TILES*TILES)  // 144 score tiles of 16x16
#define NWAVES 8
#define TPW (NTILES/NWAVES)   // 18 contiguous tiles per wave
#define TPB 256
#define SD (SDIM*DDIM)        // 12288 elements per (b,h) plane

#if __has_builtin(__builtin_amdgcn_global_load_async_to_lds_b128)
#define HAVE_ASYNC_LDS 1
#else
#define HAVE_ASYNC_LDS 0
#endif

// pointer types matching the async builtin's signature (AS1/AS3 v4i*)
typedef int v4i_vec __attribute__((vector_size(16)));
typedef __attribute__((address_space(1))) v4i_vec gv4i;
typedef __attribute__((address_space(3))) v4i_vec lv4i;

// ---- wave32 XOR butterflies via ds_swizzle_b32 (group-of-32 mode) ----
template <int MASK>
__device__ __forceinline__ float swz_add(float v) {
    int t = __builtin_amdgcn_ds_swizzle(__float_as_int(v), (MASK << 10) | 0x1F);
    return v + __int_as_float(t);
}
template <int MASK>
__device__ __forceinline__ float swz_max(float v) {
    int t = __builtin_amdgcn_ds_swizzle(__float_as_int(v), (MASK << 10) | 0x1F);
    return fmaxf(v, __int_as_float(t));
}

// ---- WMMA fragment loaders (CDNA5 wave32 layouts, ISA 7.12.2) ----
__device__ __forceinline__ v16bf load_a_frag(const __bf16* __restrict__ A,
                                             int t0, int m, int hi, int kbase) {
    const __bf16* rp = A + (t0 + m) * DDIM + kbase;
    v16bf a;
#pragma unroll
    for (int i = 0; i < 8; ++i) {
        a[i]     = rp[hi * 8 + i];
        a[i + 8] = rp[16 + hi * 8 + i];
    }
    return a;
}
__device__ __forceinline__ v16bf load_b_frag(const __bf16* __restrict__ K2,
                                             int s0, int n, int hi, int kbase) {
    const __bf16* rp = K2 + (s0 + n) * DDIM + kbase;
    v16bf b;
#pragma unroll
    for (int i = 0; i < 8; ++i) {
        b[i]     = rp[hi * 8 + i];
        b[i + 8] = rp[16 + hi * 8 + i];
    }
    return b;
}

__global__ __launch_bounds__(TPB) void trittention_kernel(
    const float* __restrict__ q,
    const float* __restrict__ k1,
    const float* __restrict__ k2,
    const float* __restrict__ v1,
    const float* __restrict__ v2,
    float* __restrict__ z_out,
    float* __restrict__ lse_out)
{
    extern __shared__ float scores[];     // [SDIM][SDIM] f32, 144 KB dynamic
    __shared__ __bf16 Abf[SD];            // (k1 * q) bf16, 24 KB
    __shared__ __bf16 Kb [SD];            // k2 bf16, 24 KB
    __shared__ float  qs[DDIM];
    __shared__ float  rowAcc[SDIM];
    __shared__ float  colAcc[SDIM];
    __shared__ float  wred[NWAVES];
    __shared__ float  sM, sSum;
    __shared__ float  zpart[4][DDIM];

    const int tid  = threadIdx.x;
    const int lane = tid & 31;
    const int wave = tid >> 5;
    const int n    = lane & 15;
    const int hi   = lane >> 4;

    const int qi = blockIdx.x;            // 0..191
    const int bh = blockIdx.y;            // 0..15
    const size_t base = (size_t)bh * SD;

    const float* k1g = k1 + base;
    const float* k2g = k2 + base;

    if (tid < DDIM) qs[tid] = q[base + (size_t)qi * DDIM + tid];
    for (int i = tid; i < SDIM; i += TPB) colAcc[i] = 0.f;

    // prefetch v1/v2 rows used by the tail GEMV (global_prefetch_b8)
    __builtin_prefetch(v1 + base + (size_t)tid * 48, 0, 1);
    __builtin_prefetch(v2 + base + (size_t)tid * 48, 0, 1);

#if HAVE_ASYNC_LDS
    // ---- stage raw f32 k1/k2 into the (not-yet-used) scores region via
    //      async global->LDS DMA (ASYNCcnt path) ----
    {
        float* k1f = scores;
        float* k2f = scores + SD;
        for (int i = tid * 4; i < SD; i += TPB * 4) {
            __builtin_amdgcn_global_load_async_to_lds_b128(
                (gv4i*)(void*)(k1g + i), (lv4i*)(k1f + i), 0, 0);
            __builtin_amdgcn_global_load_async_to_lds_b128(
                (gv4i*)(void*)(k2g + i), (lv4i*)(k2f + i), 0, 0);
        }
#if __has_builtin(__builtin_amdgcn_s_wait_asynccnt)
        __builtin_amdgcn_s_wait_asynccnt(0);
#else
        asm volatile("s_wait_asynccnt 0x0" ::: "memory");
#endif
    }
#endif
    __syncthreads();

    // ---- bf16 conversion into WMMA operand buffers ----
#if HAVE_ASYNC_LDS
    {
        const float* k1f = scores;
        const float* k2f = scores + SD;
        for (int i = tid; i < SD; i += TPB) {
            int h = i & (DDIM - 1);
            Abf[i] = (__bf16)(k1f[i] * qs[h]);
            Kb[i]  = (__bf16)(k2f[i]);
        }
    }
#else
    for (int i = tid; i < SD; i += TPB) {
        int h = i & (DDIM - 1);
        Abf[i] = (__bf16)(k1g[i] * qs[h]);
        Kb[i]  = (__bf16)(k2g[i]);
    }
#endif
    __syncthreads();

    // ---- single WMMA pass: scores = A @ k2^T, store to LDS, track max ----
    // Each wave owns 18 CONTIGUOUS tiles (<= 3 distinct tile-rows), so the A
    // fragments stay register-resident across the s0 sweep (wave-uniform
    // reload branch on t0 change).
    float lmax = -INFINITY;
    {
        int cur_t0 = -1;
        v16bf a0, a1;
        for (int tt = wave * TPW; tt < (wave + 1) * TPW; ++tt) {
            const int t0 = (tt / TILES) * 16;
            const int s0 = (tt % TILES) * 16;
            if (t0 != cur_t0) {
                a0 = load_a_frag(Abf, t0, n, hi, 0);
                a1 = load_a_frag(Abf, t0, n, hi, 32);
                cur_t0 = t0;
            }
            v16bf b0 = load_b_frag(Kb, s0, n, hi, 0);
            v16bf b1 = load_b_frag(Kb, s0, n, hi, 32);
            v8f c = {};
            c = __builtin_amdgcn_wmma_f32_16x16x32_bf16(false, a0, false, b0,
                                                        (short)0, c, false, false);
            c = __builtin_amdgcn_wmma_f32_16x16x32_bf16(false, a1, false, b1,
                                                        (short)0, c, false, false);
#pragma unroll
            for (int r = 0; r < 8; ++r) {
                scores[(t0 + hi * 8 + r) * SDIM + s0 + n] = c[r];
                lmax = fmaxf(lmax, c[r]);
            }
        }
    }
    lmax = swz_max<1>(lmax);  lmax = swz_max<2>(lmax);
    lmax = swz_max<4>(lmax);  lmax = swz_max<8>(lmax);
    lmax = swz_max<16>(lmax);
    if (lane == 0) wred[wave] = lmax;
    __syncthreads();
    if (tid == 0) {
        float mm = wred[0];
        for (int i = 1; i < NWAVES; ++i) mm = fmaxf(mm, wred[i]);
        sM = mm;
    }
    __syncthreads();
    const float m_all = sM;

    // ---- streaming softmax pass over stored scores: marginals only ----
    {
        const int t0 = wave * (SDIM / NWAVES);            // 24 exclusive rows
        float colpart[6] = {0.f, 0.f, 0.f, 0.f, 0.f, 0.f};
        for (int t = t0; t < t0 + (SDIM / NWAVES); ++t) {
            const float* srow = scores + t * SDIM;
            float rsum = 0.f;
#pragma unroll
            for (int j = 0; j < 6; ++j) {
                float p = __expf(srow[lane + 32 * j] - m_all);
                rsum += p;
                colpart[j] += p;
            }
            rsum = swz_add<1>(rsum);  rsum = swz_add<2>(rsum);
            rsum = swz_add<4>(rsum);  rsum = swz_add<8>(rsum);
            rsum = swz_add<16>(rsum);
            if (lane == 0) rowAcc[t] = rsum;              // exclusive: no atomic
        }
#pragma unroll
        for (int j = 0; j < 6; ++j)                       // 8 adders per entry
            atomicAdd(&colAcc[lane + 32 * j], colpart[j]);
    }
    __syncthreads();

    // ---- suma = sum_t rowAcc[t]  (wave-0 swizzle reduction) + lse ----
    if (wave == 0) {
        float s = 0.f;
#pragma unroll
        for (int j = 0; j < 6; ++j) s += rowAcc[lane + 32 * j];
        s = swz_add<1>(s);  s = swz_add<2>(s);
        s = swz_add<4>(s);  s = swz_add<8>(s);
        s = swz_add<16>(s);
        if (lane == 0) {
            sSum = s;
            lse_out[bh * SDIM + qi] = m_all + __logf(s);
        }
    }
    __syncthreads();
    const float suma = sSum;

    // ---- z[q,d] = (rowAcc . v1[:,d] + colAcc . v2[:,d]) / suma ----
    {
        const int d     = tid & (DDIM - 1);
        const int chunk = tid >> 6;                       // 4 chunks x 48 rows
        const float* v1p = v1 + base;
        const float* v2p = v2 + base;
        float acc = 0.f;
        for (int t = chunk * 48; t < (chunk + 1) * 48; ++t)
            acc += rowAcc[t] * v1p[t * DDIM + d] + colAcc[t] * v2p[t * DDIM + d];
        zpart[chunk][d] = acc;
    }
    __syncthreads();
    if (tid < DDIM) {
        float zt = zpart[0][tid] + zpart[1][tid] + zpart[2][tid] + zpart[3][tid];
        z_out[base + (size_t)qi * DDIM + tid] = zt / suma;
    }
}

extern "C" void kernel_launch(void* const* d_in, const int* in_sizes, int n_in,
                              void* d_out, int out_size, void* d_ws, size_t ws_size,
                              hipStream_t stream) {
    (void)in_sizes; (void)n_in; (void)out_size; (void)d_ws; (void)ws_size;
    const float* q  = (const float*)d_in[0];
    const float* k1 = (const float*)d_in[1];
    const float* k2 = (const float*)d_in[2];
    const float* v1 = (const float*)d_in[3];
    const float* v2 = (const float*)d_in[4];

    const int B = 2, H = 8;
    float* z   = (float*)d_out;                               // [B,H,S,D]
    float* lse = (float*)d_out + (size_t)B * H * SDIM * DDIM; // [B,H,S]

    const size_t dyn_lds = (size_t)SDIM * SDIM * sizeof(float);  // 144 KB
    (void)hipFuncSetAttribute((const void*)trittention_kernel,
                              hipFuncAttributeMaxDynamicSharedMemorySize,
                              (int)dyn_lds);

    dim3 grid(SDIM, B * H);   // one workgroup per (b,h,q)
    dim3 block(TPB);          // 8 wave32s
    trittention_kernel<<<grid, block, dyn_lds, stream>>>(q, k1, k2, v1, v2, z, lse);
}
